// CRF_76888504533408
// MI455X (gfx1250) — matrix operand order repaired and checked by reading, not compiled, not run on hardware
//
#include <hip/hip_runtime.h>

typedef __attribute__((ext_vector_type(16))) _Float16 v16h;
typedef __attribute__((ext_vector_type(8)))  _Float16 v8h;
typedef __attribute__((ext_vector_type(8)))  float    v8f;

#define NLAB      128
#define START_IDX 126
#define STOP_IDX  127

#define LOG2E 1.4426950408889634f
#define LN2   0.6931471805599453f

// ds_swizzle group-of-32: imm = xor<<10 | or<<5 | and  (and=0x1f -> lane^xor)
template <int IMM>
__device__ __forceinline__ float swz(float x) {
    return __builtin_bit_cast(float,
        __builtin_amdgcn_ds_swizzle(__builtin_bit_cast(int, x), IMM));
}

// 32 blocks x 256 threads (8 waves). Each block owns 16 batch rows; each wave
// owns a 16-wide label-column tile of the 128-wide alpha vector.
template <int TS>   // TS>0: compile-time T (immediate vmem offsets); TS==0: dynamic
__launch_bounds__(256, 1)
__global__ void crf_fwd_kernel(const float* __restrict__ logits,
                               const int*   __restrict__ lens,
                               const float* __restrict__ trans,
                               float* __restrict__ out,
                               int B, int Trt)
{
    const int T = (TS > 0) ? TS : Trt;

    __shared__ _Float16 Pbuf[16 * NLAB];          // P[b][k] = exp(alpha - m), f16
    __shared__ __align__(16) float wred[16 * 8];  // per-row, per-wave partials
    __shared__ float mbuf[16];
    __shared__ int   smaxlen;

    const int tid     = threadIdx.x;
    const int wave    = tid >> 5;        // 0..7  -> j-tile
    const int lane    = tid & 31;
    const int col     = lane & 15;       // N column within tile (also A-row M)
    const int half    = lane >> 4;       // 0/1
    const int rowbase = half * 8;        // C-layout rows owned: rowbase..rowbase+7
    const int j       = wave * 16 + col; // global label column
    const int b0      = blockIdx.x * 16; // batch base

    // ---- B operand: E^T tile (E = exp(Trans)), register-resident for all steps.
    // B 32x16 f16 layout: lane = column N, element e -> K = 32c + 8*half + e (e<8), 32c + 16 + 8*half + (e-8) (e>=8).
    v16h Bt[4];
    #pragma unroll
    for (int c = 0; c < 4; ++c) {
        #pragma unroll
        for (int e = 0; e < 16; ++e) {
            int k = 32 * c + (e < 8 ? 8 * half + e : 16 + 8 * half + (e - 8));
            Bt[c][e] = (_Float16)__builtin_amdgcn_exp2f(trans[j * NLAB + k] * LOG2E);
        }
    }
    const float estop = __builtin_amdgcn_exp2f(trans[STOP_IDX * NLAB + j] * LOG2E);

    // ---- per-row recurrence state (replicated across lanes/waves)
    float m_r[8];
    int   len_r[8];
    int   lmax = 1;
    #pragma unroll
    for (int r = 0; r < 8; ++r) {
        m_r[r]   = 0.0f;                 // alpha0 max = alpha0[START] = 0
        len_r[r] = lens[b0 + rowbase + r];
        lmax = max(lmax, len_r[r]);
    }

    // per-lane logit base: row r at step t lives at lgp + r*T*NLAB + t*NLAB
    const float* lgp = logits + (size_t)(b0 + rowbase) * T * NLAB + j;

    // ---- init P (one-hot at START), block max sequence length
    if (tid == 0) smaxlen = 1;
    for (int i = tid; i < 16 * NLAB; i += 256)
        Pbuf[i] = ((i & (NLAB - 1)) == START_IDX) ? (_Float16)1.0f : (_Float16)0.0f;
    __syncthreads();
    atomicMax(&smaxlen, lmax);
    __syncthreads();
    const int tmax = smaxlen;            // all rows frozen beyond this step

    for (int t = 0; t < tmax; ++t) {
        // ---- prefetch this step's logits (streamed once -> non-temporal)
        float lg[8];
        #pragma unroll
        for (int r = 0; r < 8; ++r)
            lg[r] = __builtin_nontemporal_load(lgp + (size_t)r * T * NLAB);
        lgp += NLAB;

        // ================= phase 1: Y(16x16) = P(16x128) x E^T tile ============
        v8f acc = {};
        #pragma unroll
        for (int c = 0; c < 4; ++c) {
            // A 16x32 f16 layout: row M = col; half 0 holds K 0-7 & 16-23,
            // half 1 holds K 8-15 & 24-31  -> two contiguous 16B LDS reads.
            v8h alo = *(const v8h*)&Pbuf[col * NLAB + 32 * c + 8 * half];
            v8h ahi = *(const v8h*)&Pbuf[col * NLAB + 32 * c + 16 + 8 * half];
            v16h a;
            #pragma unroll
            for (int e = 0; e < 8; ++e) { a[e] = alo[e]; a[e + 8] = ahi[e]; }
            acc = __builtin_amdgcn_wmma_f32_16x16x32_f16(false, a, false, Bt[c],
                                                         (short)0, acc, false, false);
        }

        // alpha_new[b,j] = m[b] + log(y) + logit[b,t,j]   (C row r -> b = rowbase+r)
        float anew[8], v[8];
        #pragma unroll
        for (int r = 0; r < 8; ++r) {
            float y = fmaxf(acc[r], 1e-37f);                       // guard log(0)
            anew[r] = fmaf(__builtin_amdgcn_logf(y), LN2, m_r[r]) + lg[r];
            v[r] = anew[r];
        }

        // per-row max over this wave's 16 columns: stage-outer xor-swizzle so all
        // 8 rows' DS ops pipeline per stage (stays within each 16-lane half).
        float s0[8], s1[8], s2[8], s3[8];
        #pragma unroll
        for (int r = 0; r < 8; ++r) s0[r] = swz<0x041f>(v[r]);
        #pragma unroll
        for (int r = 0; r < 8; ++r) v[r] = fmaxf(v[r], s0[r]);
        #pragma unroll
        for (int r = 0; r < 8; ++r) s1[r] = swz<0x081f>(v[r]);
        #pragma unroll
        for (int r = 0; r < 8; ++r) v[r] = fmaxf(v[r], s1[r]);
        #pragma unroll
        for (int r = 0; r < 8; ++r) s2[r] = swz<0x101f>(v[r]);
        #pragma unroll
        for (int r = 0; r < 8; ++r) v[r] = fmaxf(v[r], s2[r]);
        #pragma unroll
        for (int r = 0; r < 8; ++r) s3[r] = swz<0x201f>(v[r]);
        #pragma unroll
        for (int r = 0; r < 8; ++r) v[r] = fmaxf(v[r], s3[r]);

        if (col == 0) {
            #pragma unroll
            for (int r = 0; r < 8; ++r) wred[(rowbase + r) * 8 + wave] = v[r];
        }
        __syncthreads();   // B1: all P/m reads done; partial maxima visible

        // ================= phase 2: global max, state update ===================
        #pragma unroll
        for (int r = 0; r < 8; ++r) {
            const float4* q = (const float4*)&wred[(rowbase + r) * 8];
            float4 q0 = q[0], q1 = q[1];
            float mnew = fmaxf(fmaxf(fmaxf(q0.x, q0.y), fmaxf(q0.z, q0.w)),
                               fmaxf(fmaxf(q1.x, q1.y), fmaxf(q1.z, q1.w)));
            if (t < len_r[r]) {   // unmasked: commit; masked: freeze (skip store)
                float p = __builtin_amdgcn_exp2f((anew[r] - mnew) * LOG2E);
                Pbuf[(rowbase + r) * NLAB + j] = (_Float16)p;
                m_r[r] = mnew;
            }
        }
        __syncthreads();   // B2: new P/m committed before next step's reads
    }

    // ========== finalize: out[b] = m[b] + log( sum_j P[b,j]*exp(Trans[STOP,j]) )
    float v[8];
    #pragma unroll
    for (int r = 0; r < 8; ++r)
        v[r] = (float)Pbuf[(rowbase + r) * NLAB + j] * estop;
    #pragma unroll
    for (int r = 0; r < 8; ++r) v[r] += swz<0x041f>(v[r]);
    #pragma unroll
    for (int r = 0; r < 8; ++r) v[r] += swz<0x081f>(v[r]);
    #pragma unroll
    for (int r = 0; r < 8; ++r) v[r] += swz<0x101f>(v[r]);
    #pragma unroll
    for (int r = 0; r < 8; ++r) v[r] += swz<0x201f>(v[r]);
    if (col == 0) {
        #pragma unroll
        for (int r = 0; r < 8; ++r) {
            wred[(rowbase + r) * 8 + wave] = v[r];
            if (wave == 0) mbuf[rowbase + r] = m_r[r];
        }
    }
    __syncthreads();
    if (wave == 0 && lane < 16) {
        const float4* q = (const float4*)&wred[lane * 8];
        float4 q0 = q[0], q1 = q[1];
        float s = (q0.x + q0.y + q0.z + q0.w) + (q1.x + q1.y + q1.z + q1.w);
        out[b0 + lane] = fmaf(__builtin_amdgcn_logf(s), LN2, mbuf[lane]);
    }
}

extern "C" void kernel_launch(void* const* d_in, const int* in_sizes, int n_in,
                              void* d_out, int out_size, void* d_ws, size_t ws_size,
                              hipStream_t stream) {
    const float* logits = (const float*)d_in[0];
    const int*   lens   = (const int*)d_in[1];
    const float* trans  = (const float*)d_in[2];
    float*       out    = (float*)d_out;

    const int B = in_sizes[1];                 // 512
    const int T = in_sizes[0] / (B * NLAB);    // 512

    if (T == 512)
        crf_fwd_kernel<512><<<B / 16, 256, 0, stream>>>(logits, lens, trans, out, B, T);
    else
        crf_fwd_kernel<0><<<B / 16, 256, 0, stream>>>(logits, lens, trans, out, B, T);
}
